// LSTM_45088566673866
// MI455X (gfx1250) — matrix operand order repaired and checked by reading, not compiled, not run on hardware
//
#include <hip/hip_runtime.h>
#include <math.h>

// LSTM, 2 layers, T=1024, B=16, I=H=1024. WMMA f16 x f16 -> f32 path.

typedef __attribute__((ext_vector_type(16))) _Float16 v16h;
typedef __attribute__((ext_vector_type(8)))  _Float16 v8h;
typedef __attribute__((ext_vector_type(8)))  float    v8f;
typedef __attribute__((ext_vector_type(4)))  int      v4i;

// Pointer-to-AS(1)/AS(3) v4i, matching __builtin_amdgcn_global_load_async_to_lds_b128.
typedef __attribute__((address_space(1))) v4i* gptr_v4i;
typedef __attribute__((address_space(3))) v4i* lptr_v4i;

#define TDIM 1024
#define BDIM 16
#define HDIM 1024
#define GDIM 4096   // 4*H

// ---------- WMMA fragment loaders (CDNA5 16x16x32 f16 layouts, ISA 7.12.2) ----

// A-matrix 16x32 f16: lane<16 -> row=lane,    K = kq..kq+7, kq+16..kq+23 (kq=0)
//                     lane>=16 -> row=lane-16, same with kq=8
__device__ __forceinline__ v16h load_fragA(const _Float16* __restrict__ base,
                                           int m0, int ldk, int k0, int lane) {
  int r  = lane & 15;
  int kq = (lane >> 4) * 8;
  const _Float16* p = base + (size_t)(m0 + r) * ldk + (k0 + kq);
  v8h lo = *(const v8h*)(p);
  v8h hi = *(const v8h*)(p + 16);
  v16h a;
#pragma unroll
  for (int i = 0; i < 8; ++i) { a[i] = lo[i]; a[i + 8] = hi[i]; }
  return a;
}

// B-matrix 32x16 f16: lane holds column n0+(lane%16), K = kq..kq+7, kq+16..kq+23
// with kq=(lane/16)*8. W is row-major [N,K] so a B column == a W row (contiguous).
__device__ __forceinline__ v16h load_fragB(const _Float16* __restrict__ w,
                                           int n0, int ldk, int k0, int lane) {
  int c  = lane & 15;
  int kq = (lane >> 4) * 8;
  const _Float16* p = w + (size_t)(n0 + c) * ldk + (k0 + kq);
  v8h lo = *(const v8h*)(p);
  v8h hi = *(const v8h*)(p + 16);
  v16h b;
#pragma unroll
  for (int i = 0; i < 8; ++i) { b[i] = lo[i]; b[i + 8] = hi[i]; }
  return b;
}

// ---------- f32 -> f16 conversion ------------------------------------------

__global__ void lstm_f32_to_f16(const float* __restrict__ in,
                                _Float16* __restrict__ out, size_t n) {
  size_t i = (size_t)blockIdx.x * blockDim.x + threadIdx.x;
  size_t stride = (size_t)gridDim.x * blockDim.x;
  for (; i < n; i += stride) out[i] = (_Float16)in[i];
}

// ---------- Input-projection GEMM: C[M,N] = A[M,K] * W[N,K]^T + b0[N]+b1[N] --
// 256 threads = 8 waves arranged 2(M) x 4(N); each wave register-blocks a
// 64x64 macro-tile = 4x4 WMMA tiles (16 wmma per 8 fragment loads per K-step).
// Block tile = 128(M) x 256(N). Grid = (M/128, N/256).

__global__ __launch_bounds__(256) void lstm_gemm_wmma_bias(
    const _Float16* __restrict__ A,     // [M, K]
    const _Float16* __restrict__ W,     // [N, K]
    const float*    __restrict__ bias0, // [N]
    const float*    __restrict__ bias1, // [N]
    float*          __restrict__ C,     // [M, N]
    int M, int N, int K) {
  int lane  = threadIdx.x & 31;
  int wave  = threadIdx.x >> 5;
  int waveM = wave >> 2;                      // 0..1
  int waveN = wave & 3;                       // 0..3
  int m0 = blockIdx.x * 128 + waveM * 64;
  int n0 = blockIdx.y * 256 + waveN * 64;

  v8f acc[4][4] = {};
  for (int k0 = 0; k0 < K; k0 += 32) {
    v16h a[4], b[4];
#pragma unroll
    for (int i = 0; i < 4; ++i) a[i] = load_fragA(A, m0 + i * 16, K, k0, lane);
#pragma unroll
    for (int j = 0; j < 4; ++j) b[j] = load_fragB(W, n0 + j * 16, K, k0, lane);
#pragma unroll
    for (int i = 0; i < 4; ++i)
#pragma unroll
      for (int j = 0; j < 4; ++j)
        acc[i][j] = __builtin_amdgcn_wmma_f32_16x16x32_f16(
            false, a[i], false, b[j], (short)0, acc[i][j], false, false);
  }

  int cr = lane >> 4;   // 0/1 -> row halves
  int cc = lane & 15;   // column within tile
#pragma unroll
  for (int j = 0; j < 4; ++j) {
    int n = n0 + j * 16 + cc;
    float bsum = bias0[n] + bias1[n];
#pragma unroll
    for (int i = 0; i < 4; ++i) {
#pragma unroll
      for (int v = 0; v < 8; ++v) {
        int m = m0 + i * 16 + v + cr * 8;
        C[(size_t)m * N + n] = acc[i][j][v] + bsum;
      }
    }
  }
}

// ---------- Fused recurrent step --------------------------------------------
// gates[b, n(+gH)] = gx[t][b, n(+gH)] + sum_k h[b,k] * Whh[n(+gH), k]
// hprev (16x1024 f16 = 32KB) is staged into LDS once per block via async
// global->LDS copies (ASYNCcnt), then every wave reads A-fragments from LDS.
// One wave owns 16 columns of H and computes all 4 gate tiles, then the
// c/h elementwise update. 64 waves total = 8 blocks x 8 waves.

__global__ __launch_bounds__(256) void lstm_step_wmma(
    const float*    __restrict__ gx,     // [B, 4H] slice for t (biases folded)
    const _Float16* __restrict__ Whh,    // [4H, H] f16
    const _Float16* __restrict__ hprev,  // [B, H] f16 (history slot t)
    float*          __restrict__ c,      // [B, H] f32 state
    float*          __restrict__ hOutF32,// [B, H] f32 (layer output slice)
    _Float16*       __restrict__ hOutF16)// [B, H] f16 (history slot t+1)
{
  __shared__ __align__(16) _Float16 hsh[BDIM * HDIM];   // 32KB

  int tid  = threadIdx.x;
  int lane = tid & 31;
  int wave = tid >> 5;

  // Stage hprev into LDS: 2048 x 16B chunks, 8 per thread.
#if __has_builtin(__builtin_amdgcn_global_load_async_to_lds_b128)
#pragma unroll
  for (int i = 0; i < 8; ++i) {
    int idx = (tid + i * 256) * 8;    // in halves; 16B aligned
    __builtin_amdgcn_global_load_async_to_lds_b128(
        (gptr_v4i)(hprev + idx),      // global src: v4i in addrspace(1)
        (lptr_v4i)(&hsh[idx]),        // lds dst:    v4i in addrspace(3)
        0, 0);
  }
#if __has_builtin(__builtin_amdgcn_s_wait_asynccnt)
  __builtin_amdgcn_s_wait_asynccnt(0);
#else
  asm volatile("s_wait_asynccnt 0" ::: "memory");
#endif
#else
#pragma unroll
  for (int i = 0; i < 8; ++i) {
    int idx = (tid + i * 256) * 8;
    *(v8h*)(&hsh[idx]) = *(const v8h*)(hprev + idx);
  }
#endif
  __syncthreads();

  int nb = (blockIdx.x * 8 + wave) * 16;   // column base in [0, H)

  v8f acc[4] = {};
  for (int k0 = 0; k0 < HDIM; k0 += 32) {
    v16h a = load_fragA(hsh, 0, HDIM, k0, lane);    // A from LDS (ds_load)
#pragma unroll
    for (int g = 0; g < 4; ++g) {
      v16h b = load_fragB(Whh, g * HDIM + nb, HDIM, k0, lane);
      acc[g] = __builtin_amdgcn_wmma_f32_16x16x32_f16(
          false, a, false, b, (short)0, acc[g], false, false);
    }
  }

  int cr = lane >> 4;
  int col = nb + (lane & 15);
#pragma unroll
  for (int v = 0; v < 8; ++v) {
    int b = v + cr * 8;                    // batch row
    const float* gxb = gx + (size_t)b * GDIM + col;
    float gi = acc[0][v] + gxb[0];
    float gf = acc[1][v] + gxb[HDIM];
    float gg = acc[2][v] + gxb[2 * HDIM];
    float go = acc[3][v] + gxb[3 * HDIM];
    float si = 1.0f / (1.0f + __expf(-gi));
    float sf = 1.0f / (1.0f + __expf(-gf));
    float tg = tanhf(gg);
    float so = 1.0f / (1.0f + __expf(-go));
    size_t idx = (size_t)b * HDIM + col;
    float cn = sf * c[idx] + si * tg;
    float hn = so * tanhf(cn);
    c[idx] = cn;
    hOutF32[idx] = hn;
    hOutF16[idx] = (_Float16)hn;
  }
}

// ---------- host ------------------------------------------------------------

extern "C" void kernel_launch(void* const* d_in, const int* in_sizes, int n_in,
                              void* d_out, int out_size, void* d_ws, size_t ws_size,
                              hipStream_t stream) {
  (void)in_sizes; (void)n_in; (void)out_size; (void)ws_size;

  const float* x     = (const float*)d_in[0];
  const float* Wih0  = (const float*)d_in[1];
  const float* bih0  = (const float*)d_in[2];
  const float* Whh0  = (const float*)d_in[3];
  const float* bhh0  = (const float*)d_in[4];
  const float* Wih1  = (const float*)d_in[5];
  const float* bih1  = (const float*)d_in[6];
  const float* Whh1  = (const float*)d_in[7];
  const float* bhh1  = (const float*)d_in[8];
  float* out = (float*)d_out;

  const size_t M   = (size_t)TDIM * BDIM;          // 16384 rows for big GEMMs
  const size_t TBH = (size_t)TDIM * BDIM * HDIM;   // 16777216

  // Workspace carve-up (256B aligned).
  char* ws = (char*)d_ws;
  size_t off = 0;
  auto carve = [&](size_t bytes) -> void* {
    void* p = ws + off;
    off = (off + bytes + 255) & ~(size_t)255;
    return p;
  };
  _Float16* xf16   = (_Float16*)carve(M * HDIM * sizeof(_Float16));           // 32MB
  _Float16* wih0f  = (_Float16*)carve((size_t)GDIM * HDIM * sizeof(_Float16));// 8MB
  _Float16* whh0f  = (_Float16*)carve((size_t)GDIM * HDIM * sizeof(_Float16));
  _Float16* wih1f  = (_Float16*)carve((size_t)GDIM * HDIM * sizeof(_Float16));
  _Float16* whh1f  = (_Float16*)carve((size_t)GDIM * HDIM * sizeof(_Float16));
  float*    gx     = (float*)carve(M * GDIM * sizeof(float));                 // 256MB
  _Float16* h0hist = (_Float16*)carve((size_t)(TDIM + 1) * BDIM * HDIM * sizeof(_Float16));
  _Float16* h1hist = (_Float16*)carve((size_t)(TDIM + 1) * BDIM * HDIM * sizeof(_Float16));
  float*    c0     = (float*)carve((size_t)BDIM * HDIM * sizeof(float));
  float*    c1     = (float*)carve((size_t)BDIM * HDIM * sizeof(float));
  float*    hcur0  = (float*)carve((size_t)BDIM * HDIM * sizeof(float));

  // 1) f32 -> f16 conversions.
  lstm_f32_to_f16<<<2048, 256, 0, stream>>>(x,    xf16,  M * HDIM);
  lstm_f32_to_f16<<<1024, 256, 0, stream>>>(Wih0, wih0f, (size_t)GDIM * HDIM);
  lstm_f32_to_f16<<<1024, 256, 0, stream>>>(Whh0, whh0f, (size_t)GDIM * HDIM);
  lstm_f32_to_f16<<<1024, 256, 0, stream>>>(Wih1, wih1f, (size_t)GDIM * HDIM);
  lstm_f32_to_f16<<<1024, 256, 0, stream>>>(Whh1, whh1f, (size_t)GDIM * HDIM);

  dim3 gemmGrid(M / 128, GDIM / 256);

  // ---------------- Layer 0 ----------------
  lstm_gemm_wmma_bias<<<gemmGrid, 256, 0, stream>>>(
      xf16, wih0f, bih0, bhh0, gx, (int)M, GDIM, HDIM);
  (void)hipMemsetAsync(h0hist, 0, (size_t)BDIM * HDIM * sizeof(_Float16), stream);
  (void)hipMemsetAsync(c0,     0, (size_t)BDIM * HDIM * sizeof(float),    stream);
  for (int t = 0; t < TDIM; ++t) {
    lstm_step_wmma<<<8, 256, 0, stream>>>(
        gx + (size_t)t * BDIM * GDIM, whh0f,
        h0hist + (size_t)t * BDIM * HDIM, c0,
        hcur0,                                   // only last step's h is needed in f32
        h0hist + (size_t)(t + 1) * BDIM * HDIM);
  }

  // ---------------- Layer 1 ----------------
  lstm_gemm_wmma_bias<<<gemmGrid, 256, 0, stream>>>(
      h0hist + (size_t)BDIM * HDIM,   // slots 1..T == layer-0 outputs, f16
      wih1f, bih1, bhh1, gx, (int)M, GDIM, HDIM);
  (void)hipMemsetAsync(h1hist, 0, (size_t)BDIM * HDIM * sizeof(_Float16), stream);
  (void)hipMemsetAsync(c1,     0, (size_t)BDIM * HDIM * sizeof(float),    stream);
  for (int t = 0; t < TDIM; ++t) {
    lstm_step_wmma<<<8, 256, 0, stream>>>(
        gx + (size_t)t * BDIM * GDIM, whh1f,
        h1hist + (size_t)t * BDIM * HDIM, c1,
        out + (size_t)t * BDIM * HDIM,           // layer-1 output -> d_out
        h1hist + (size_t)(t + 1) * BDIM * HDIM);
  }

  // Tail outputs: next_h [2,B,H] then next_c [2,B,H] appended after out1.
  const size_t BH = (size_t)BDIM * HDIM;
  (void)hipMemcpyAsync(out + TBH,          hcur0,            sizeof(float) * BH,
                       hipMemcpyDeviceToDevice, stream);
  (void)hipMemcpyAsync(out + TBH + BH,     out + (TBH - BH), sizeof(float) * BH,
                       hipMemcpyDeviceToDevice, stream);
  (void)hipMemcpyAsync(out + TBH + 2 * BH, c0,               sizeof(float) * BH,
                       hipMemcpyDeviceToDevice, stream);
  (void)hipMemcpyAsync(out + TBH + 3 * BH, c1,               sizeof(float) * BH,
                       hipMemcpyDeviceToDevice, stream);
}
